// QwenBlockCache_75015898792030
// MI455X (gfx1250) — matrix operand order repaired and checked by reading, not compiled, not run on hardware
//
#include <hip/hip_runtime.h>
#include <hip/hip_bf16.h>
#include <stdint.h>

// ---------------------------------------------------------------------------
// Qwen decode block for MI455X (gfx1250, wave32, WMMA + TDM).
// Memory-bound (~1.9 GB traffic -> ~81us floor @ 23.3 TB/s). Weight streaming
// uses the Tensor Data Mover (tensor_load_to_lds) with double-buffered LDS
// tiles; matrix math is v_wmma_f32_16x16x32_f16 on padded 16x16 tiles (M=8).
// ---------------------------------------------------------------------------

typedef __attribute__((ext_vector_type(16))) _Float16 v16h;
typedef __attribute__((ext_vector_type(2)))  _Float16 v2h;
typedef __attribute__((ext_vector_type(8)))  float    v8f;
typedef __attribute__((ext_vector_type(4)))  uint32_t v4u;
typedef __attribute__((ext_vector_type(8)))  int      v8i;
typedef __attribute__((ext_vector_type(4)))  int      v4i;

#define HIDDEN   4096
#define HEADS    32
#define HEAD_DIM 128
#define FF       11008
#define CACHE    2048
#define KVLEN    (CACHE + 1)      // 2049
#define BATCH    8
#define QKV3     (3 * HIDDEN)     // 12288
#define WPITCH   68               // fp32 LDS row pitch: 64 + 4 pad DWORDs
                                  // (TDM pad_interval=64dw, pad_amount=4dw)
#define TILEFP   (32 * WPITCH)    // one 32-row fp32 tile in LDS (floats)

#if __has_builtin(__builtin_amdgcn_tensor_load_to_lds) && \
    __has_builtin(__builtin_amdgcn_s_wait_tensorcnt)
#define USE_TDM 1
#endif

union ABPack { uint4 u[2]; v16h h; };

static __device__ __forceinline__ uint32_t pack_f16x2(float a, float b) {
  // single v_cvt_pk_rtz_f16_f32
  return __builtin_bit_cast(uint32_t, __builtin_amdgcn_cvt_pkrtz(a, b));
}

// ---------------------------------------------------------------------------
// RMSNorm -> f16-pair activation matrix [16][K/2], rows 8..15 zero (M padding)
// ---------------------------------------------------------------------------
__global__ void rmsnorm_pack_kernel(const float* __restrict__ x,
                                    const float* __restrict__ w,
                                    uint32_t* __restrict__ outp) {
  const int m = blockIdx.x;            // 0..15
  const int tid = threadIdx.x;         // 256
  const int KP = HIDDEN / 2;
  if (m >= BATCH) {
    for (int i = tid; i < KP; i += 256) outp[(size_t)m * KP + i] = 0u;
    return;
  }
  __shared__ float red[256];
  const float* xr = x + (size_t)m * HIDDEN;
  float s = 0.f;
  for (int i = tid; i < HIDDEN; i += 256) { float v = xr[i]; s += v * v; }
  red[tid] = s; __syncthreads();
  for (int off = 128; off > 0; off >>= 1) {
    if (tid < off) red[tid] += red[tid + off];
    __syncthreads();
  }
  const float scale = rsqrtf(red[0] / (float)HIDDEN + 1e-6f);
  for (int j = tid; j < KP; j += 256) {
    float a = xr[2 * j]     * scale * w[2 * j];
    float b = xr[2 * j + 1] * scale * w[2 * j + 1];
    outp[(size_t)m * KP + j] = pack_f16x2(a, b);
  }
}

__global__ void zero_u32_kernel(uint32_t* __restrict__ p, int n) {
  int i = blockIdx.x * blockDim.x + threadIdx.x;
  const int stride = gridDim.x * blockDim.x;
  for (; i < n; i += stride) p[i] = 0u;
}

// ---------------------------------------------------------------------------
// TDM: DMA one 32x64 fp32 weight tile (row stride K) into LDS with a 4-DWORD
// pad after every 64 DWORDs (-> pitch WPITCH, bank-conflict-free ds_load_b128).
// D# per CDNA5 ISA ch.8: group0 {count=1, lds_addr, global_addr, type=2},
// group1 {data_size=4B, pad_en, pad_interval=5(64dw), pad_amount=3(4dw),
//         tensor_dim0=K, tensor_dim1=32, tile_dim0=64, tile_dim1=32,
//         tensor_dim0_stride=K}, groups 2/3 zero (2-D tile).
// This toolchain exposes the 6-arg builtin form:
//   (uint32x4 g0, int32x8 g1, int32x4 g2, int32x4 g3, int32x8 extra, i32 cpol)
// ---------------------------------------------------------------------------
#ifdef USE_TDM
static __device__ __forceinline__ void tdm_issue_tile(const float* W, int n0,
                                                      int K, int kb,
                                                      const float* ldsbuf) {
  const uint64_t ga = (uint64_t)(uintptr_t)(W + (size_t)n0 * K + kb);
  const uint32_t lb = (uint32_t)(uintptr_t)ldsbuf;  // low 32b = LDS byte offset
  v4u g0; g0[0] = 1u; g0[1] = lb; g0[2] = (uint32_t)ga;
  g0[3] = (uint32_t)((ga >> 32) & 0x01FFFFFFu) | (2u << 30);
  v8i g1;
  g1[0] = (int)((2u << 16) | (1u << 20) | (5u << 22) | (3u << 25));
  g1[1] = (int)(((uint32_t)K & 0xFFFFu) << 16);           // tensor_dim0 lo
  g1[2] = (int)((((uint32_t)K >> 16) & 0xFFFFu) | (32u << 16)); // dim0 hi|dim1
  g1[3] = (int)(64u << 16);                               // tile_dim0 = 64
  g1[4] = 32;                                             // tile_dim1 = 32
  g1[5] = K;                                              // dim0 stride (lo32)
  g1[6] = 0; g1[7] = 0;
  v4i gz; gz[0] = 0; gz[1] = 0; gz[2] = 0; gz[3] = 0;
  v8i gz8; gz8[0] = 0; gz8[1] = 0; gz8[2] = 0; gz8[3] = 0;
  gz8[4] = 0; gz8[5] = 0; gz8[6] = 0; gz8[7] = 0;
  __builtin_amdgcn_tensor_load_to_lds(g0, g1, gz, gz, gz8, 0);
}
#endif

// ---------------------------------------------------------------------------
// Core: one 32-row weight tile x full K, 2 waves per block, 16 rows per wave.
// LDS holds raw fp32 tiles (pitch WPITCH); A operand converted lane-side with
// v_cvt_pk_rtz into the 16x32 f16 A-stripe:
//   lane<16: row m, K {0..7,16..23}; lane>=16: row m, K {8..15,24..31}.
// B (32x16 f16, from L2-resident activation pairs): column = lane&15,
//   lane<16 K-pairs {0..3,8..11}, lane>=16 K-pairs {4..7,12..15}.
// ---------------------------------------------------------------------------
static __device__ __forceinline__ void gemm_tile_core(
    const float* __restrict__ W,      // row-major [N][K]
    int n0,                           // base row of this block's 32-row tile
    int K,                            // reduction length in floats (mult of 64)
    const uint32_t* __restrict__ Bp,  // activations^T as f16 pairs [16][K/2]
    float* lds,                       // [2][32][WPITCH] fp32
    v8f& acc) {
  const int tid  = threadIdx.x;       // 64
  const int wave = tid >> 5;
  const int lane = tid & 31;
  const int m    = lane & 15;
  const int half = lane >> 4;
  const int KP   = K >> 1;

#ifdef USE_TDM
  if (wave == 0) tdm_issue_tile(W, n0, K, 0, lds);          // prime buffer 0
#else
  const int srow = tid >> 4;          // fallback staging row (0..3)
  const int sc4  = tid & 15;          // fallback staging float4 column
#endif

  for (int kb = 0; kb < K; kb += 64) {
    const int buf = (kb >> 6) & 1;
    float* cur = lds + buf * TILEFP;
#ifdef USE_TDM
    // issue next tile into the other buffer, then wait for current tile.
    if (wave == 0) {
      if (kb + 64 < K) {
        tdm_issue_tile(W, n0, K, kb + 64, lds + (buf ^ 1) * TILEFP);
        __builtin_amdgcn_s_wait_tensorcnt(1);
      } else {
        __builtin_amdgcn_s_wait_tensorcnt(0);
      }
    }
    __syncthreads();
#else
    // manual staging fallback: coalesced b128 copy of 32x64 fp32 tile
#pragma unroll
    for (int sw = 0; sw < 8; ++sw) {
      const int r = sw * 4 + srow;
      const float* gp = W + (size_t)(n0 + r) * K + kb + sc4 * 4;
      *(float4*)(cur + r * WPITCH + sc4 * 4) = *(const float4*)gp;
      if (kb + 64 < K) __builtin_prefetch(gp + 64, 0, 3);
    }
    __syncthreads();
#endif
    // ---- two 16x16x32 WMMA steps over the staged 64-K block ----
#pragma unroll
    for (int ks = 0; ks < 2; ++ks) {
      const float* ar = cur + (wave * 16 + m) * WPITCH + ks * 32 + half * 8;
      const float4 f0 = *(const float4*)(ar);
      const float4 f1 = *(const float4*)(ar + 4);
      const float4 f2 = *(const float4*)(ar + 16);
      const float4 f3 = *(const float4*)(ar + 20);
      ABPack a, b;
      a.u[0] = make_uint4(pack_f16x2(f0.x, f0.y), pack_f16x2(f0.z, f0.w),
                          pack_f16x2(f1.x, f1.y), pack_f16x2(f1.z, f1.w));
      a.u[1] = make_uint4(pack_f16x2(f2.x, f2.y), pack_f16x2(f2.z, f2.w),
                          pack_f16x2(f3.x, f3.y), pack_f16x2(f3.z, f3.w));
      const uint32_t* bb = Bp + (size_t)m * KP + (kb >> 1) + ks * 16 + half * 4;
      b.u[0] = *(const uint4*)(bb);
      b.u[1] = *(const uint4*)(bb + 8);
      acc = __builtin_amdgcn_wmma_f32_16x16x32_f16(
          false, a.h, false, b.h, (short)0, acc, false, false);
    }
    __syncthreads();
  }
}

// mode 0: out = A*W1^T + bias      (qkv)            outF stride = QKV3
// mode 1: out = resid + A*W1^T    (attn proj -> h)  outF stride = HIDDEN
// mode 2: pairs = (A*W1^T) * silu(A*W2^T)  -> outP [16][FF/2]
// mode 3: out = resid + A*W1^T    (mlp proj -> out) outF stride = HIDDEN
__global__ void gemm_wmma_kernel(const float* __restrict__ W1,
                                 const float* __restrict__ W2,
                                 int K,
                                 const uint32_t* __restrict__ Bp,
                                 const float* __restrict__ bias,
                                 const float* __restrict__ resid,
                                 float* __restrict__ outF,
                                 uint32_t* __restrict__ outP,
                                 int outStride, int mode) {
  __shared__ float lds[2 * TILEFP];
  const int tid  = threadIdx.x;
  const int wave = tid >> 5;
  const int lane = tid & 31;
  const int m    = lane & 15;
  const int half = lane >> 4;
  const int n0   = blockIdx.x * 32;
  const int n0w  = n0 + wave * 16 + half * 8;   // D layout: VGPR r -> row n0w+r

  v8f acc = {};
  gemm_tile_core(W1, n0, K, Bp, lds, acc);

  if (mode == 2) {
    v8f acc2 = {};
    __syncthreads();
    gemm_tile_core(W2, n0, K, Bp, lds, acc2);
    if (m < BATCH) {
#pragma unroll
      for (int i = 0; i < 4; ++i) {
        float g0 = acc2[2 * i], g1 = acc2[2 * i + 1];
        float s0 = acc[2 * i]     * (g0 / (1.f + __expf(-g0)));
        float s1 = acc[2 * i + 1] * (g1 / (1.f + __expf(-g1)));
        outP[(size_t)m * (FF / 2) + (n0w >> 1) + i] = pack_f16x2(s0, s1);
      }
    }
    return;
  }
  if (m < BATCH) {
#pragma unroll
    for (int r = 0; r < 8; ++r) {
      const int n = n0w + r;
      float v = acc[r];
      if (mode == 0) v += bias[n];
      else           v += resid[(size_t)m * outStride + n];
      outF[(size_t)m * outStride + n] = v;
    }
  }
}

// ---------------------------------------------------------------------------
// RoPE + split qkv; new k,v written into slot CACHE of k_full/v_full outputs.
// ---------------------------------------------------------------------------
__global__ void rope_split_kernel(const float* __restrict__ qkv,
                                  const int* __restrict__ pos_ids,
                                  const float* __restrict__ cosT,
                                  const float* __restrict__ sinT,
                                  float* __restrict__ qout,
                                  float* __restrict__ kcache,
                                  float* __restrict__ vcache) {
  const int b = blockIdx.x >> 5;
  const int h = blockIdx.x & 31;
  const int d = threadIdx.x;                       // 128
  const size_t qb = (size_t)b * QKV3 + h * HEAD_DIM;
  const float q = qkv[qb + d];
  const float k = qkv[qb + HIDDEN + d];
  const float v = qkv[qb + 2 * HIDDEN + d];
  const int pos = pos_ids[b];
  const float c = cosT[(size_t)pos * HEAD_DIM + d];
  const float s = sinT[(size_t)pos * HEAD_DIM + d];
  float qr, kr;
  if (d < 64) { qr = -qkv[qb + d + 64];  kr = -qkv[qb + HIDDEN + d + 64]; }
  else        { qr =  qkv[qb + d - 64];  kr =  qkv[qb + HIDDEN + d - 64]; }
  qout[((size_t)b * HEADS + h) * HEAD_DIM + d] = q * c + qr * s;
  const size_t co = (((size_t)b * KVLEN + CACHE) * HEADS + h) * HEAD_DIM + d;
  kcache[co] = k * c + kr * s;
  vcache[co] = v;
}

// ---------------------------------------------------------------------------
// Attention per (b,h): softmax over 2049 keys; KV-cache copy fused into the
// two passes so past_k/past_v are each read exactly once, written once.
// ---------------------------------------------------------------------------
__global__ void attention_kernel(const float* __restrict__ pastk,
                                 const float* __restrict__ pastv,
                                 const float* __restrict__ mask,   // [8][2049]
                                 const float* __restrict__ qvecs,  // [8][32][128]
                                 float* __restrict__ kcache,       // [8][2049][32][128]
                                 float* __restrict__ vcache,
                                 uint32_t* __restrict__ ctxp) {    // [16][2048] pairs
  const int b = blockIdx.x >> 5;
  const int h = blockIdx.x & 31;
  const int tid  = threadIdx.x;   // 256 = 8 waves
  const int wave = tid >> 5;
  const int lane = tid & 31;

  __shared__ float sc[KVLEN];
  __shared__ float red[256];
  __shared__ float qv[HEAD_DIM];
  __shared__ float part[8 * HEAD_DIM];
  __shared__ float ctx[HEAD_DIM];

  if (tid < HEAD_DIM) qv[tid] = qvecs[((size_t)b * HEADS + h) * HEAD_DIM + tid];
  __syncthreads();
  const float4 q4 = *(const float4*)(qv + lane * 4);
  const float scale = 0.08838834764831845f;        // 1/sqrt(128)

  // pass 1: scores, fused copy past_k -> k_full
  for (int k = wave; k < KVLEN; k += 8) {
    const size_t dsto = (((size_t)b * KVLEN + k) * HEADS + h) * HEAD_DIM + lane * 4;
    float4 k4;
    if (k < CACHE) {
      k4 = *(const float4*)(pastk + (((size_t)b * CACHE + k) * HEADS + h) * HEAD_DIM + lane * 4);
      *(float4*)(kcache + dsto) = k4;
    } else {
      k4 = *(const float4*)(kcache + dsto);        // new token (RoPE output)
    }
    float dot = q4.x * k4.x + q4.y * k4.y + q4.z * k4.z + q4.w * k4.w;
#pragma unroll
    for (int off = 16; off > 0; off >>= 1) dot += __shfl_xor(dot, off, 32);
    if (lane == 0) sc[k] = dot * scale + mask[(size_t)b * KVLEN + k];
  }
  __syncthreads();

  // softmax over 2049 scores
  float lm = -1e30f;
  for (int k = tid; k < KVLEN; k += 256) lm = fmaxf(lm, sc[k]);
  red[tid] = lm; __syncthreads();
  for (int off = 128; off > 0; off >>= 1) {
    if (tid < off) red[tid] = fmaxf(red[tid], red[tid + off]);
    __syncthreads();
  }
  const float mx = red[0]; __syncthreads();
  float ls = 0.f;
  for (int k = tid; k < KVLEN; k += 256) { float e = __expf(sc[k] - mx); sc[k] = e; ls += e; }
  red[tid] = ls; __syncthreads();
  for (int off = 128; off > 0; off >>= 1) {
    if (tid < off) red[tid] += red[tid + off];
    __syncthreads();
  }
  const float inv = 1.f / red[0];
  __syncthreads();

  // pass 2: ctx = sum p*v, fused copy past_v -> v_full
  float4 a4 = make_float4(0.f, 0.f, 0.f, 0.f);
  for (int k = wave; k < KVLEN; k += 8) {
    const size_t dsto = (((size_t)b * KVLEN + k) * HEADS + h) * HEAD_DIM + lane * 4;
    float4 v4;
    if (k < CACHE) {
      v4 = *(const float4*)(pastv + (((size_t)b * CACHE + k) * HEADS + h) * HEAD_DIM + lane * 4);
      *(float4*)(vcache + dsto) = v4;
    } else {
      v4 = *(const float4*)(vcache + dsto);
    }
    const float p = sc[k];
    a4.x += p * v4.x; a4.y += p * v4.y; a4.z += p * v4.z; a4.w += p * v4.w;
  }
  *(float4*)(part + wave * HEAD_DIM + lane * 4) = a4;
  __syncthreads();
  if (tid < HEAD_DIM) {
    float s = 0.f;
#pragma unroll
    for (int w = 0; w < 8; ++w) s += part[w * HEAD_DIM + tid];
    ctx[tid] = s * inv;
  }
  __syncthreads();
  if (tid < 64) {   // pack ctx row (K-pairs for next GEMM's B operand)
    ctxp[(size_t)b * (HIDDEN / 2) + h * (HEAD_DIM / 2) + tid] =
        pack_f16x2(ctx[2 * tid], ctx[2 * tid + 1]);
  }
}

// ---------------------------------------------------------------------------
extern "C" void kernel_launch(void* const* d_in, const int* in_sizes, int n_in,
                              void* d_out, int out_size, void* d_ws, size_t ws_size,
                              hipStream_t stream) {
  const float* hidden = (const float*)d_in[0];
  const int*   pos    = (const int*)  d_in[1];
  const float* mask   = (const float*)d_in[2];
  const float* pastk  = (const float*)d_in[3];
  const float* pastv  = (const float*)d_in[4];
  const float* ln1w   = (const float*)d_in[5];
  const float* ln2w   = (const float*)d_in[6];
  const float* wqkv   = (const float*)d_in[7];
  const float* bqkv   = (const float*)d_in[8];
  const float* wattn  = (const float*)d_in[9];
  const float* w1     = (const float*)d_in[10];
  const float* w2     = (const float*)d_in[11];
  const float* wmlp   = (const float*)d_in[12];
  const float* cosT   = (const float*)d_in[13];
  const float* sinT   = (const float*)d_in[14];

  float* out   = (float*)d_out;                        // [8][4096]
  float* kfull = out + (size_t)BATCH * HIDDEN;         // [8][2049][32][128]
  float* vfull = kfull + (size_t)BATCH * KVLEN * HIDDEN;

  uint32_t* ws32 = (uint32_t*)d_ws;
  uint32_t* ln1p = ws32;                               // 16*2048
  uint32_t* ln2p = ln1p + 16 * 2048;
  uint32_t* ctxp = ln2p + 16 * 2048;
  uint32_t* actp = ctxp + 16 * 2048;                   // 16*5504
  float* qkvws = (float*)(actp + 16 * 5504);           // 8*12288
  float* qws   = qkvws + (size_t)BATCH * QKV3;         // 8*4096
  float* hws   = qws + (size_t)BATCH * HIDDEN;         // 8*4096

  rmsnorm_pack_kernel<<<16, 256, 0, stream>>>(hidden, ln1w, ln1p);
  zero_u32_kernel<<<64, 256, 0, stream>>>(ctxp + 8 * 2048, 8 * 2048);
  zero_u32_kernel<<<64, 256, 0, stream>>>(actp + 8 * 5504, 8 * 5504);

  gemm_wmma_kernel<<<QKV3 / 32, 64, 0, stream>>>(
      wqkv, nullptr, HIDDEN, ln1p, bqkv, nullptr, qkvws, nullptr, QKV3, 0);

  rope_split_kernel<<<256, 128, 0, stream>>>(qkvws, pos, cosT, sinT, qws, kfull, vfull);
  attention_kernel<<<256, 256, 0, stream>>>(pastk, pastv, mask, qws, kfull, vfull, ctxp);

  gemm_wmma_kernel<<<HIDDEN / 32, 64, 0, stream>>>(
      wattn, nullptr, HIDDEN, ctxp, nullptr, hidden, hws, nullptr, HIDDEN, 1);

  rmsnorm_pack_kernel<<<16, 256, 0, stream>>>(hws, ln2w, ln2p);

  gemm_wmma_kernel<<<FF / 32, 64, 0, stream>>>(
      w1, w2, HIDDEN, ln2p, nullptr, nullptr, nullptr, actp, 0, 2);

  gemm_wmma_kernel<<<HIDDEN / 32, 64, 0, stream>>>(
      wmlp, nullptr, FF, actp, nullptr, hws, out, nullptr, HIDDEN, 3);

  (void)in_sizes; (void)n_in; (void)out_size; (void)ws_size;
}
